// RNN_13383118094552
// MI455X (gfx1250) — matrix operand order repaired and checked by reading, not compiled, not run on hardware
//
#include <hip/hip_runtime.h>
#include <hip/hip_bf16.h>

#define T_STEPS 2048
#define BATCH   64
#define IDIM    512
#define HDIM    512
#define NKB     (HDIM / 32)   // 16 K-blocks of 32
#define NWG_SCAN 32           // 512 output cols / 16 per WG

typedef __attribute__((ext_vector_type(16))) __bf16 v16bf;
typedef __attribute__((ext_vector_type(8)))  float  v8f;

// ============================================================================
// Fragment-major layouts (CDNA5 WMMA 16x16x32 bf16, wave32, ISA 7.12.2):
//  B frag buffer:  Wf[((nb*NKB + kb)*32 + lane)*16 + j]
//      = W[kb*32 + ((lane<16)?0:8) + j + ((j>=8)?8:0)][nb*16 + (lane&15)]
//  A frag buffer:  Hf[((mt*NKB + kb)*32 + lane)*16 + j]
//      = H[mt*16 + (lane&15)][kb*32 + ((lane<16)?0:8) + j + ((j>=8)?8:0)]
// Every lane slice is 16 contiguous bf16 (32 B, 32 B aligned) -> 2x b128 loads.
// ============================================================================

__device__ inline v16bf load_frag(const __bf16* __restrict__ F, int blk, int kb,
                                  int lane) {
  const size_t idx = ((((size_t)blk * NKB + kb) * 32) + lane) << 4;
  return *(const v16bf*)(F + idx);
}

// A (16x32 bf16) from an f32 row-major source, converting on the fly (xproj).
__device__ inline v16bf load_a_f32(const float* __restrict__ src, int row, int ld,
                                   int k0, int lane) {
  const int kb = (lane < 16) ? 0 : 8;
  const float* p0 = src + (size_t)row * ld + k0 + kb;
  const float* p1 = p0 + 16;
  v16bf a;
#pragma unroll
  for (int j = 0; j < 8; ++j) {
    a[j]     = (__bf16)p0[j];
    a[8 + j] = (__bf16)p1[j];
  }
  return a;
}

// ---------- prep: f32 weights -> bf16, permuted into fragment-major layout ----
__global__ void cvt_weights(const float* __restrict__ w,
                            __bf16* __restrict__ wxf, __bf16* __restrict__ whf) {
  const int i = blockIdx.x * blockDim.x + threadIdx.x;
  if (i >= IDIM * HDIM) return;
  const int j    = i & 15;
  const int lane = (i >> 4) & 31;
  const int kb   = (i >> 9) & (NKB - 1);
  const int nb   = i >> 13;
  const int k = kb * 32 + ((lane < 16) ? 0 : 8) + j + ((j >= 8) ? 8 : 0);
  const int n = nb * 16 + (lane & 15);
  wxf[i] = (__bf16)w[(size_t)k * HDIM + n];            // Wx rows 0..511
  whf[i] = (__bf16)w[(size_t)(IDIM + k) * HDIM + n];   // Wh rows 512..1023
}

// ------ prep: out[0] = broadcast(h0); seed A-frag parity-0 buffer; zero sync --
__global__ void init_h0(float* __restrict__ out, const float* __restrict__ h0,
                        __bf16* __restrict__ hf0, unsigned* __restrict__ sync) {
  const int i = blockIdx.x * blockDim.x + threadIdx.x;
  if (i < 2) sync[i] = 0u;
  if (i >= BATCH * HDIM) return;
  out[i] = h0[i & (HDIM - 1)];
  // A-frag layout index -> K (value independent of M: h0 is broadcast)
  const int j    = i & 15;
  const int lane = (i >> 4) & 31;
  const int kb   = (i >> 9) & (NKB - 1);
  const int k = kb * 32 + ((lane < 16) ? 0 : 8) + j + ((j >= 8) ? 8 : 0);
  hf0[i] = (__bf16)h0[k];
}

// ---------- parallel GEMM: out[t+1] = X_t @ Wx + bias (pre-activation) -------
__global__ void __launch_bounds__(128)
xproj_gemm(const float* __restrict__ X, const __bf16* __restrict__ Wxf,
           const float* __restrict__ bias, float* __restrict__ out) {
  const int lane = threadIdx.x & 31;
  const int wave = threadIdx.x >> 5;
  const int m0  = blockIdx.x * 64 + wave * 16;   // row in flattened (T*B)
  const int nb0 = blockIdx.y * 4;                // 4 N-blocks of 16
  const int mh  = (lane < 16) ? 0 : 8;
  const int nl  = lane & 15;

  v8f c[4];
#pragma unroll
  for (int j = 0; j < 4; ++j) {
    const float bv = bias[(nb0 + j) * 16 + nl];
#pragma unroll
    for (int r = 0; r < 8; ++r) c[j][r] = bv;
  }

  for (int kb = 0; kb < IDIM / 32; ++kb) {
    const v16bf a = load_a_f32(X, m0 + (lane & 15), IDIM, kb * 32, lane);
#pragma unroll
    for (int j = 0; j < 4; ++j) {
      const v16bf b = load_frag(Wxf, nb0 + j, kb, lane);
      c[j] = __builtin_amdgcn_wmma_f32_16x16x32_bf16(
                 false, a, false, b, (short)0, c[j], false, false);
    }
  }

  float* dst = out + (size_t)BATCH * HDIM;  // time rows start at t=1
#pragma unroll
  for (int j = 0; j < 4; ++j)
#pragma unroll
    for (int r = 0; r < 8; ++r)
      dst[(size_t)(m0 + r + mh) * HDIM + (nb0 + j) * 16 + nl] = c[j][r];
}

// ---------- sequential scan: out[t+1] = tanh(out[t+1] + out[t] @ Wh) ---------
// 32 persistent WGs x 4 waves; Wh panel resident in VGPRs for all 2048 steps.
// Hidden state ping-pongs between two bf16 A-fragment buffers: reads are pure
// b128 loads, no per-step f32->bf16 conversion on the critical path.
__global__ void __launch_bounds__(128)
rnn_scan(float* __restrict__ out, const __bf16* __restrict__ Whf,
         __bf16* __restrict__ hfrag, unsigned* __restrict__ sync) {
  const int lane = threadIdx.x & 31;
  const int wave = threadIdx.x >> 5;
  const int nb = blockIdx.x;
  const int n0 = nb * 16;
  const int m0 = wave * 16;
  const int mh = (lane < 16) ? 0 : 8;
  const int nl = lane & 15;
  unsigned* cnt = sync;
  unsigned* gen = sync + 1;

  // this thread's scatter coordinates in the A-frag buffer (n acts as next K)
  const int n    = n0 + nl;
  const int kbW  = n >> 5;
  const int kpr  = n & 31;
  const int lhW  = (kpr & 8) ? 16 : 0;
  const int jW   = (kpr & 7) + ((kpr & 16) ? 8 : 0);
  const size_t scat_base = (((size_t)wave * NKB + kbW) * 32 + lhW + mh) * 16 + jW;

  v16bf bh[NKB];
#pragma unroll
  for (int kb = 0; kb < NKB; ++kb) bh[kb] = load_frag(Whf, nb, kb, lane);

  for (int t = 0; t < T_STEPS; ++t) {
    float* hn = out + (size_t)(t + 1) * BATCH * HDIM;   // xproj_t -> h_{t+1}
    const __bf16* hA  = hfrag + (size_t)(t & 1) * BATCH * HDIM;
    __bf16*       hAn = hfrag + (size_t)((t + 1) & 1) * BATCH * HDIM;

    // accumulator init = xproj + bias already sitting in out[t+1]
    v8f c;
#pragma unroll
    for (int r = 0; r < 8; ++r)
      c[r] = hn[(size_t)(m0 + r + mh) * HDIM + n0 + nl];

    // warm next step's pre-activation row while the WMMA chain runs
    if (t + 2 <= T_STEPS)
      __builtin_prefetch(out + (size_t)(t + 2) * BATCH * HDIM +
                             (size_t)(m0 + (lane & 15)) * HDIM + n0, 0, 1);

#pragma unroll
    for (int kb = 0; kb < NKB; ++kb) {
      const v16bf a = load_frag(hA, wave, kb, lane);
      c = __builtin_amdgcn_wmma_f32_16x16x32_bf16(
              false, a, false, bh[kb], (short)0, c, false, false);
    }

#pragma unroll
    for (int r = 0; r < 8; ++r) {
      const float v = tanhf(c[r]);
      hn[(size_t)(m0 + r + mh) * HDIM + n0 + nl] = v;   // f32 output
      hAn[scat_base + (size_t)r * 16] = (__bf16)v;      // bf16 A-frag for t+1
    }

    // device-wide barrier between time steps (persistent-kernel sync)
    __syncthreads();
    if (threadIdx.x == 0) {
      __threadfence();
      const unsigned target = (unsigned)(t + 1) * NWG_SCAN;
      if (atomicAdd(cnt, 1u) == target - 1u)
        __hip_atomic_store(gen, (unsigned)(t + 1), __ATOMIC_RELEASE,
                           __HIP_MEMORY_SCOPE_AGENT);
      while (__hip_atomic_load(gen, __ATOMIC_ACQUIRE, __HIP_MEMORY_SCOPE_AGENT) <
             (unsigned)(t + 1))
        __builtin_amdgcn_s_sleep(2);
    }
    __syncthreads();
  }
}

extern "C" void kernel_launch(void* const* d_in, const int* in_sizes, int n_in,
                              void* d_out, int out_size, void* d_ws, size_t ws_size,
                              hipStream_t stream) {
  (void)in_sizes; (void)n_in; (void)out_size; (void)ws_size;
  const float* X    = (const float*)d_in[0];
  const float* W    = (const float*)d_in[1];
  const float* bias = (const float*)d_in[2];
  const float* h0   = (const float*)d_in[3];
  float* out = (float*)d_out;

  char* ws = (char*)d_ws;
  __bf16*   wxf   = (__bf16*)ws;                                      // 512 KB
  __bf16*   whf   = (__bf16*)(ws + (size_t)IDIM * HDIM * 2);          // 512 KB
  __bf16*   hfrag = (__bf16*)(ws + (size_t)IDIM * HDIM * 4);          // 2 x 64 KB
  unsigned* sync  = (unsigned*)(ws + (size_t)IDIM * HDIM * 4 +
                                (size_t)2 * BATCH * HDIM * 2);

  cvt_weights<<<(IDIM * HDIM + 255) / 256, 256, 0, stream>>>(W, wxf, whf);
  init_h0<<<(BATCH * HDIM + 255) / 256, 256, 0, stream>>>(out, h0, hfrag, sync);

  dim3 g(T_STEPS * BATCH / 64, HDIM / 64);
  xproj_gemm<<<g, 128, 0, stream>>>(X, wxf, bias, out);

  rnn_scan<<<NWG_SCAN, 128, 0, stream>>>(out, whf, hfrag, sync);
}
